// AttentionSeq2Seq_6262062318267
// MI455X (gfx1250) — compile-verified, matrix operand order checked by reading, not run on hardware
//
#include <hip/hip_runtime.h>
#include <hip/hip_bf16.h>
#include <math.h>

typedef __attribute__((ext_vector_type(16))) _Float16 v16h;
typedef __attribute__((ext_vector_type(8)))  float    v8f;

#define B_    1024
#define H_    256
#define G3_   768
#define KIN_  32
#define KOUT_ 16
#define ML_   80
#define LPRE_ 60
#define LFWD_ 80

// ---------------------------------------------------------------------------
// Fragment layout helpers (CDNA5 16-bit WMMA A/B operand layout, §7.12.2):
// element (row, k):  lane = (row&15) + 16*g,  half-index j inside v16h,
// where within k2 = k%32:
//   k2 in [0,8)   -> g=0, j=k2        k2 in [8,16)  -> g=1, j=k2-8
//   k2 in [16,24) -> g=0, j=k2-8      k2 in [24,32) -> g=1, j=k2-16
// Fragments stored as [rowTile][ktBase+kt][lane(32)][j(16)] halves so each
// lane's fragment is one contiguous 32-byte v16h load.
// ---------------------------------------------------------------------------
__device__ __forceinline__ void frag_store(_Float16* dst, int ktStride, int ktBase,
                                           int row, int k, _Float16 v) {
  int mt = row >> 4, lr = row & 15;
  int kt = k >> 5,  k2 = k & 31;
  int g, j;
  if      (k2 < 8)  { g = 0; j = k2;      }
  else if (k2 < 16) { g = 1; j = k2 - 8;  }
  else if (k2 < 24) { g = 0; j = k2 - 8;  }
  else              { g = 1; j = k2 - 16; }
  dst[(((size_t)mt * ktStride + ktBase + kt) * 32 + lr + 16 * g) * 16 + j] = v;
}

// Generic f32 [rows,K] row-major -> fragment pack (weights and activations).
__global__ __launch_bounds__(256) void pack_frag_kernel(
    const float* __restrict__ src, _Float16* __restrict__ dst,
    int K, int ktStride, int ktBase) {
  int idx = blockIdx.x * 256 + threadIdx.x;
  int row = idx / K, k = idx - row * K;
  frag_store(dst, ktStride, ktBase, row, k, (_Float16)src[idx]);
}

__global__ __launch_bounds__(256) void zero_u32_kernel(unsigned int* p, int n) {
  int i = blockIdx.x * 256 + threadIdx.x;
  if (i < n) p[i] = 0u;
}

// ---------------------------------------------------------------------------
// WMMA GEMM:  C[1024, N] = act( A[1024, 32*KT] * W^T + bias )
// 2x2 register blocking: each wave owns a 32x32 C block (4 accumulators),
// 4 wmma per 4 fragment loads (1:1 vmem:wmma in the steady-state loop).
// 8 waves/block -> 128x64 C per block.  grid = (1024/128, ceil(Ntiles/4)).
// Edge columns (Ntiles % 2 == 1) take a uniform-branch 2x1 loop so EXEC
// stays all-ones around every WMMA.
// ---------------------------------------------------------------------------
__device__ __forceinline__ v8f wmma_f16(v16h a, v16h b, v8f c) {
  return __builtin_amdgcn_wmma_f32_16x16x32_f16(false, a, false, b,
                                                (short)0, c, false, false);
}

__global__ __launch_bounds__(256) void gemm_frag_kernel(
    const _Float16* __restrict__ A, int aKtStride, int aKtBase,
    const _Float16* __restrict__ Bw,
    const float* __restrict__ bias,
    float* __restrict__ C,
    int Ntiles, int KT, int ldc, int act) {
  int lane = threadIdx.x & 31;
  int wave = threadIdx.x >> 5;
  int mt0 = blockIdx.x * 8 + (wave & 3) * 2;            // 2 M-tiles per wave
  int nt0 = (blockIdx.y * 2 + (wave >> 2)) * 2;         // 2 N-tiles per wave
  if (nt0 >= Ntiles) return;                            // wave-uniform
  bool n1ok = (nt0 + 1) < Ntiles;

  v8f c00 = {}, c01 = {}, c10 = {}, c11 = {};
  const _Float16* ap0 = A + ((size_t)(mt0 * aKtStride + aKtBase) * 32 + lane) * 16;
  const _Float16* ap1 = ap0 + (size_t)aKtStride * 512;  // next M-tile
  const _Float16* bp0 = Bw + ((size_t)(nt0 * KT) * 32 + lane) * 16;
  const _Float16* bp1 = bp0 + (size_t)KT * 512;         // next N-tile

  if (n1ok) {
    for (int kt = 0; kt < KT; ++kt) {
      v16h a0 = *(const v16h*)ap0; ap0 += 512;
      v16h a1 = *(const v16h*)ap1; ap1 += 512;
      v16h b0 = *(const v16h*)bp0; bp0 += 512;
      v16h b1 = *(const v16h*)bp1; bp1 += 512;
      c00 = wmma_f16(a0, b0, c00);
      c01 = wmma_f16(a0, b1, c01);
      c10 = wmma_f16(a1, b0, c10);
      c11 = wmma_f16(a1, b1, c11);
    }
  } else {
    for (int kt = 0; kt < KT; ++kt) {
      v16h a0 = *(const v16h*)ap0; ap0 += 512;
      v16h a1 = *(const v16h*)ap1; ap1 += 512;
      v16h b0 = *(const v16h*)bp0; bp0 += 512;
      c00 = wmma_f16(a0, b0, c00);
      c10 = wmma_f16(a1, b0, c10);
    }
  }

  int colL = lane & 15, rowL = 8 * (lane >> 4);
  auto store_tile = [&](v8f cc, int mt, int nt) {
    int col   = nt * 16 + colL;
    int rbase = mt * 16 + rowL;
    float bv = bias ? bias[col] : 0.f;
#pragma unroll
    for (int r = 0; r < 8; ++r) {
      float v = cc[r] + bv;
      if (act == 1) v = tanhf(v);
      C[(size_t)(rbase + r) * ldc + col] = v;
    }
  };
  store_tile(c00, mt0,     nt0);
  store_tile(c10, mt0 + 1, nt0);
  if (n1ok) {
    store_tile(c01, mt0,     nt0 + 1);
    store_tile(c11, mt0 + 1, nt0 + 1);
  }
}

// ---------------------------------------------------------------------------
// Encoder embedding: emb = [pre_x, pre_y] @ enc_emb_W^T + b  (K=48, tiny),
// written directly as packed x-fragment for the GRU input GEMM.
// ---------------------------------------------------------------------------
__global__ __launch_bounds__(256) void enc_embpack_kernel(
    const float* __restrict__ pre_x, const float* __restrict__ pre_y,
    const float* __restrict__ W, const float* __restrict__ bias,
    _Float16* __restrict__ xfrag, int t) {
  int idx = blockIdx.x * 256 + threadIdx.x;   // m*256 + col
  int m = idx >> 8, col = idx & 255;
  const float* xr = pre_x + ((size_t)t * B_ + m) * KIN_;
  const float* yr = pre_y + ((size_t)t * B_ + m) * KOUT_;
  const float* wr = W + col * (KIN_ + KOUT_);
  float acc = bias[col];
#pragma unroll
  for (int j = 0; j < KIN_;  ++j) acc += xr[j] * wr[j];
#pragma unroll
  for (int j = 0; j < KOUT_; ++j) acc += yr[j] * wr[KIN_ + j];
  frag_store(xfrag, 8, 0, m, col, (_Float16)acc);
}

// Decoder embedding: e = [y_prev, fx] @ dec_emb_W^T + b, packed into
// bufA kt0-7 (attention input [e,h]) and bufB kt0-7 (combine input [e,ctx]).
__global__ __launch_bounds__(256) void dec_embpack_kernel(
    const float* __restrict__ y_prev, const float* __restrict__ fx,
    const float* __restrict__ W, const float* __restrict__ bias,
    _Float16* __restrict__ bufA, _Float16* __restrict__ bufB) {
  int idx = blockIdx.x * 256 + threadIdx.x;
  int m = idx >> 8, col = idx & 255;
  const float* wr = W + col * (KOUT_ + KIN_);
  float acc = bias[col];
  if (y_prev) {
    const float* yr = y_prev + (size_t)m * KOUT_;
#pragma unroll
    for (int j = 0; j < KOUT_; ++j) acc += yr[j] * wr[j];
  }
  const float* xr = fx + (size_t)m * KIN_;
#pragma unroll
  for (int j = 0; j < KIN_; ++j) acc += xr[j] * wr[KOUT_ + j];
  _Float16 h = (_Float16)acc;
  frag_store(bufA, 16, 0, m, col, h);
  frag_store(bufB, 16, 0, m, col, h);
}

// ---------------------------------------------------------------------------
// GRU gates: r,z,n + new h. Reads gi/gh [1024,768]; updates h f32 in place,
// fuses packing of h2 into a fragment slot; optionally stores f16 encoder out.
// ---------------------------------------------------------------------------
__global__ __launch_bounds__(256) void gru_gate_kernel(
    const float* __restrict__ gi, const float* __restrict__ gh,
    const float* __restrict__ bih, const float* __restrict__ bhh,
    float* __restrict__ h, _Float16* __restrict__ hfrag,
    int ktStride, int ktBase, _Float16* __restrict__ enc_out_t) {
  int idx = blockIdx.x * 256 + threadIdx.x;
  int m = idx >> 8, col = idx & 255;
  const float* gir = gi + (size_t)m * G3_;
  const float* ghr = gh + (size_t)m * G3_;
  float ir = gir[col]       + bih[col];
  float iz = gir[col + 256] + bih[col + 256];
  float in = gir[col + 512] + bih[col + 512];
  float hr = ghr[col]       + bhh[col];
  float hz = ghr[col + 256] + bhh[col + 256];
  float hn = ghr[col + 512] + bhh[col + 512];
  float r = 1.f / (1.f + expf(-(ir + hr)));
  float z = 1.f / (1.f + expf(-(iz + hz)));
  float n = tanhf(in + r * hn);
  float h2 = (1.f - z) * n + z * h[idx];
  h[idx] = h2;
  frag_store(hfrag, ktStride, ktBase, m, col, (_Float16)h2);
  if (enc_out_t) enc_out_t[idx] = (_Float16)h2;
}

// ---------------------------------------------------------------------------
// Softmax over ML=80 logits + context ctx[b,:] = sum_m aw[m]*enc[m,b,:].
// Encoder outputs beyond L_PRE are exact zeros -> sum only m<60.
// Packs ctx into bufB kt8-15. One block (128 threads) per batch element.
// ---------------------------------------------------------------------------
__global__ __launch_bounds__(128) void softmax_ctx_kernel(
    const float* __restrict__ logits, const float* __restrict__ attn_b,
    const _Float16* __restrict__ enc_out, _Float16* __restrict__ bufB) {
  __shared__ float red[128];
  __shared__ float aw[ML_];
  int b = blockIdx.x, tid = threadIdx.x;
  float v = -1e30f;
  if (tid < ML_) v = logits[(size_t)b * ML_ + tid] + attn_b[tid];
  red[tid] = v; __syncthreads();
  for (int s = 64; s > 0; s >>= 1) {
    if (tid < s) red[tid] = fmaxf(red[tid], red[tid + s]);
    __syncthreads();
  }
  float mx = red[0]; __syncthreads();
  float e = (tid < ML_) ? expf(v - mx) : 0.f;
  red[tid] = e; __syncthreads();
  for (int s = 64; s > 0; s >>= 1) {
    if (tid < s) red[tid] += red[tid + s];
    __syncthreads();
  }
  float inv = 1.f / red[0];
  if (tid < ML_) aw[tid] = e * inv;
  __syncthreads();
  for (int col = tid; col < H_; col += 128) {
    float acc = 0.f;
    for (int m = 0; m < LPRE_; ++m)
      acc += aw[m] * (float)enc_out[((size_t)m * B_ + b) * H_ + col];
    frag_store(bufB, 16, 8, b, col, (_Float16)acc);
  }
}

// ---------------------------------------------------------------------------
extern "C" void kernel_launch(void* const* d_in, const int* in_sizes, int n_in,
                              void* d_out, int out_size, void* d_ws, size_t ws_size,
                              hipStream_t stream) {
  (void)in_sizes; (void)n_in; (void)out_size; (void)ws_size;
  const float* pre_x     = (const float*)d_in[0];
  const float* pre_y     = (const float*)d_in[1];
  const float* forward_x = (const float*)d_in[2];
  const float* enc_emb_W = (const float*)d_in[3];
  const float* enc_emb_b = (const float*)d_in[4];
  const float* enc_Wih   = (const float*)d_in[5];
  const float* enc_Whh   = (const float*)d_in[6];
  const float* enc_bih   = (const float*)d_in[7];
  const float* enc_bhh   = (const float*)d_in[8];
  const float* dec_emb_W = (const float*)d_in[9];
  const float* dec_emb_b = (const float*)d_in[10];
  const float* attn_W    = (const float*)d_in[11];
  const float* attn_b    = (const float*)d_in[12];
  const float* comb_W    = (const float*)d_in[13];
  const float* comb_b    = (const float*)d_in[14];
  const float* dec_Wih   = (const float*)d_in[15];
  const float* dec_Whh   = (const float*)d_in[16];
  const float* dec_bih   = (const float*)d_in[17];
  const float* dec_bhh   = (const float*)d_in[18];
  const float* out_W     = (const float*)d_in[19];
  const float* out_b     = (const float*)d_in[20];
  float* out = (float*)d_out;

  // Deterministic workspace carving (256B aligned).
  char* p = (char*)d_ws;
  auto carve = [&](size_t bytes) -> char* {
    char* r = p; p += (bytes + 255) & ~(size_t)255; return r;
  };
  _Float16* fWencIH = (_Float16*)carve((size_t)G3_ * H_ * 2);
  _Float16* fWencHH = (_Float16*)carve((size_t)G3_ * H_ * 2);
  _Float16* fWdecIH = (_Float16*)carve((size_t)G3_ * H_ * 2);
  _Float16* fWdecHH = (_Float16*)carve((size_t)G3_ * H_ * 2);
  _Float16* fWattn  = (_Float16*)carve((size_t)ML_ * 2 * H_ * 2);
  _Float16* fWcomb  = (_Float16*)carve((size_t)H_ * 2 * H_ * 2);
  _Float16* fWout   = (_Float16*)carve((size_t)KOUT_ * H_ * 2);
  _Float16* xfragE  = (_Float16*)carve((size_t)B_ * H_ * 2);       // enc emb frag
  _Float16* hfragE  = (_Float16*)carve((size_t)B_ * H_ * 2);       // enc h frag
  _Float16* bufA    = (_Float16*)carve((size_t)B_ * 2 * H_ * 2);   // [e|h] frag
  _Float16* bufB    = (_Float16*)carve((size_t)B_ * 2 * H_ * 2);   // [e|ctx] frag
  _Float16* gfrag   = (_Float16*)carve((size_t)B_ * H_ * 2);       // g frag
  _Float16* enc_out = (_Float16*)carve((size_t)LPRE_ * B_ * H_ * 2);
  float* h      = (float*)carve((size_t)B_ * H_ * 4);
  float* gi     = (float*)carve((size_t)B_ * G3_ * 4);
  float* gh     = (float*)carve((size_t)B_ * G3_ * 4);
  float* logits = (float*)carve((size_t)B_ * ML_ * 4);
  float* gtmp   = (float*)carve((size_t)B_ * H_ * 4);

  // --- weight packing (once per call, recomputed deterministically) ---
  pack_frag_kernel<<<G3_ * H_ / 256, 256, 0, stream>>>(enc_Wih, fWencIH, H_, 8, 0);
  pack_frag_kernel<<<G3_ * H_ / 256, 256, 0, stream>>>(enc_Whh, fWencHH, H_, 8, 0);
  pack_frag_kernel<<<G3_ * H_ / 256, 256, 0, stream>>>(dec_Wih, fWdecIH, H_, 8, 0);
  pack_frag_kernel<<<G3_ * H_ / 256, 256, 0, stream>>>(dec_Whh, fWdecHH, H_, 8, 0);
  pack_frag_kernel<<<ML_ * 2 * H_ / 256, 256, 0, stream>>>(attn_W, fWattn, 2 * H_, 16, 0);
  pack_frag_kernel<<<H_ * 2 * H_ / 256, 256, 0, stream>>>(comb_W, fWcomb, 2 * H_, 16, 0);
  pack_frag_kernel<<<KOUT_ * H_ / 256, 256, 0, stream>>>(out_W, fWout, H_, 8, 0);

  // --- initial state h = 0 (f32 and fragment form) ---
  zero_u32_kernel<<<(B_ * H_) / 256, 256, 0, stream>>>((unsigned int*)h, B_ * H_);
  zero_u32_kernel<<<(B_ * H_ / 2) / 256, 256, 0, stream>>>((unsigned int*)hfragE, B_ * H_ / 2);

  // --- encoder scan ---
  for (int t = 0; t < LPRE_; ++t) {
    enc_embpack_kernel<<<B_, 256, 0, stream>>>(pre_x, pre_y, enc_emb_W, enc_emb_b, xfragE, t);
    gemm_frag_kernel<<<dim3(8, 12), 256, 0, stream>>>(xfragE, 8, 0, fWencIH, nullptr, gi, 48, 8, G3_, 0);
    gemm_frag_kernel<<<dim3(8, 12), 256, 0, stream>>>(hfragE, 8, 0, fWencHH, nullptr, gh, 48, 8, G3_, 0);
    gru_gate_kernel<<<B_, 256, 0, stream>>>(gi, gh, enc_bih, enc_bhh, h, hfragE, 8, 0,
                                            enc_out + (size_t)t * B_ * H_);
  }
  // hand h off to decoder: pack into bufA kt8-15 ( [e|h] concat slot )
  pack_frag_kernel<<<B_ * H_ / 256, 256, 0, stream>>>(h, bufA, H_, 16, 8);

  // --- decoder scan ---
  for (int t = 0; t < LFWD_; ++t) {
    const float* yprev = t ? (out + (size_t)(t - 1) * B_ * KOUT_) : nullptr;
    dec_embpack_kernel<<<B_, 256, 0, stream>>>(yprev, forward_x + (size_t)t * B_ * KIN_,
                                               dec_emb_W, dec_emb_b, bufA, bufB);
    // logits = [e,h] @ attn_W^T    (N=80 -> 5 tiles)
    gemm_frag_kernel<<<dim3(8, 2), 256, 0, stream>>>(bufA, 16, 0, fWattn, nullptr, logits, 5, 16, ML_, 0);
    softmax_ctx_kernel<<<B_, 128, 0, stream>>>(logits, attn_b, enc_out, bufB);
    // g = tanh([e,ctx] @ comb_W^T + comb_b)
    gemm_frag_kernel<<<dim3(8, 4), 256, 0, stream>>>(bufB, 16, 0, fWcomb, comb_b, gtmp, 16, 16, H_, 1);
    pack_frag_kernel<<<B_ * H_ / 256, 256, 0, stream>>>(gtmp, gfrag, H_, 8, 0);
    // GRU gates
    gemm_frag_kernel<<<dim3(8, 12), 256, 0, stream>>>(gfrag, 8, 0, fWdecIH, nullptr, gi, 48, 8, G3_, 0);
    gemm_frag_kernel<<<dim3(8, 12), 256, 0, stream>>>(bufA, 16, 8, fWdecHH, nullptr, gh, 48, 8, G3_, 0);
    gru_gate_kernel<<<B_, 256, 0, stream>>>(gi, gh, dec_bih, dec_bhh, h, bufA, 16, 8, nullptr);
    // y = h2 @ out_W^T + out_b  (N=16 -> 1 tile)
    gemm_frag_kernel<<<dim3(8, 1), 256, 0, stream>>>(bufA, 16, 8, fWout, out_b,
                                                     out + (size_t)t * B_ * KOUT_, 1, 8, KOUT_, 0);
  }
}
